// GNN_43868795961671
// MI455X (gfx1250) — compile-verified, matrix operand order checked by reading, not run on hardware
//
#include <hip/hip_runtime.h>
#include <hip/hip_bf16.h>
#include <math.h>

// ---------------------------------------------------------------------------
// GatedGCN pipeline for MI455X (gfx1250, wave32, WMMA bf16 -> f32 accumulate)
// Activations live in bf16 (WMMA's native input); accumulators/stats in f32.
// ---------------------------------------------------------------------------

typedef __bf16 bf16_t;
typedef bf16_t v16bf __attribute__((ext_vector_type(16)));
typedef bf16_t v8bf  __attribute__((ext_vector_type(8)));
typedef float  v8f   __attribute__((ext_vector_type(8)));

#define N_NODES 50000
#define N_EDGES 1200000
#define DIM 64
#define EPS_BN 1e-5f

static_assert(N_EDGES % 128 == 0, "edge grid must tile exactly (no guards)");
static_assert(N_NODES % 16 == 0, "node tiles are all-or-nothing per wave");

__device__ __forceinline__ float sigf(float x)  { return 1.0f / (1.0f + __expf(-x)); }
__device__ __forceinline__ float siluf(float x) { return x * sigf(x); }

// f32 -> bf16 round-to-nearest-even
__device__ __forceinline__ bf16_t f2bf(float f) {
    unsigned u = __builtin_bit_cast(unsigned, f);
    u = (u + 0x7FFFu + ((u >> 16) & 1u)) >> 16;
    unsigned short s = (unsigned short)u;
    return __builtin_bit_cast(bf16_t, s);
}
__device__ __forceinline__ float bf2f(bf16_t x) { return (float)x; }

// packed-pair helpers: one u32 holds {hi=elem1, lo=elem0} bf16
__device__ __forceinline__ float bflo(unsigned u) { return __builtin_bit_cast(float, u << 16); }
__device__ __forceinline__ float bfhi(unsigned u) { return __builtin_bit_cast(float, u & 0xFFFF0000u); }
__device__ __forceinline__ unsigned pack_bf2(float a, float b) {   // a->lo, b->hi
    unsigned ua = __builtin_bit_cast(unsigned, a);
    unsigned ub = __builtin_bit_cast(unsigned, b);
    ua += 0x7FFFu + ((ua >> 16) & 1u);
    ub += 0x7FFFu + ((ub >> 16) & 1u);
    return __builtin_amdgcn_perm(ub, ua, 0x07060302);              // v_perm_b32
}

// ---- WMMA fragment loaders (wave32 layouts per CDNA5 ISA 7.12.2) ----------
// A (16x32 bf16): lane l holds row m=l&15; half (l>>4) selects K-octets:
//   elems 0..7 -> K = kh+0..7, elems 8..15 -> K = kh+16..23, kh=(l>>4)*8.
__device__ __forceinline__ v16bf load_a_bf(const bf16_t* A, long lda, int m0, int kbase) {
    int lane = threadIdx.x & 31;
    int m = m0 + (lane & 15);
    int kh = (lane >> 4) * 8;
    const bf16_t* r = A + (long)m * lda + kbase + kh;
    v8bf lo = *(const v8bf*)r;              // K = kh + 0..7
    v8bf hi = *(const v8bf*)(r + 16);       // K = kh + 16..23
    return __builtin_shufflevector(lo, hi, 0, 1, 2, 3, 4, 5, 6, 7,
                                           8, 9, 10, 11, 12, 13, 14, 15);
}

// B fragments pre-swizzled: slot t = nt*64 + ks*32 + lane holds 16 contiguous bf16.
__device__ __forceinline__ v16bf load_b_swz(const bf16_t* Wswz, int nt, int ks) {
    int lane = threadIdx.x & 31;
    return *(const v16bf*)(Wswz + (((nt * 2 + ks) * 32) + lane) * 16);
}

// One wave: D(16x64) += A(16x64) * W(64x64); acc = 4 n-tiles of 16x16 f32.
__device__ __forceinline__ void gemm16x64(const bf16_t* A, long lda, int m0,
                                          const bf16_t* Wswz, v8f acc[4]) {
    v16bf a0 = load_a_bf(A, lda, m0, 0);
    v16bf a1 = load_a_bf(A, lda, m0, 32);
#pragma unroll
    for (int nt = 0; nt < 4; ++nt) {
        acc[nt] = __builtin_amdgcn_wmma_f32_16x16x32_bf16(
            false, a0, false, load_b_swz(Wswz, nt, 0), (short)0, acc[nt], false, false);
        acc[nt] = __builtin_amdgcn_wmma_f32_16x16x32_bf16(
            false, a1, false, load_b_swz(Wswz, nt, 1), (short)0, acc[nt], false, false);
    }
}

// ---------------------------------------------------------------------------
// Weight conversion: f32 [64][64] (k-major) -> bf16 in B-fragment order.
// ---------------------------------------------------------------------------
__global__ void k_cvt_swz(const float* __restrict__ W, bf16_t* __restrict__ out) {
    int t = threadIdx.x;                 // 0..255
    int lane = t & 31, ks = (t >> 5) & 1, nt = t >> 6;
    int n  = nt * 16 + (lane & 15);
    int kh = (lane >> 4) * 8;
    bf16_t* o = out + t * 16;
#pragma unroll
    for (int j = 0; j < 8; ++j) {
        o[j]     = f2bf(W[(ks * 32 + kh + j) * DIM + n]);
        o[j + 8] = f2bf(W[(ks * 32 + 16 + kh + j) * DIM + n]);
    }
}

// ---------------------------------------------------------------------------
// Input projections (bf16 outputs; edge side packs pairs)
// ---------------------------------------------------------------------------
__global__ void k_init_h(const float* __restrict__ x, const float* __restrict__ W,
                         const float* __restrict__ b, bf16_t* __restrict__ h) {
    size_t idx = (size_t)blockIdx.x * blockDim.x + threadIdx.x;
    if (idx >= (size_t)N_NODES * DIM) return;
    int row = (int)(idx >> 6), c = (int)(idx & 63);
    float v = x[row * 2 + 0] * W[c] + x[row * 2 + 1] * W[64 + c] + b[c];
    h[idx] = f2bf(siluf(v));
}

__global__ void k_init_e(const float* __restrict__ ea, const float* __restrict__ W,
                         const float* __restrict__ b, unsigned* __restrict__ e2) {
    size_t p = (size_t)blockIdx.x * blockDim.x + threadIdx.x;   // pair index
    if (p >= (size_t)N_EDGES * DIM / 2) return;
    size_t i = p * 2;
    int row = (int)(i >> 6), c = (int)(i & 63);
    float a = ea[row];
    e2[p] = pack_bf2(siluf(a * W[c] + b[c]), siluf(a * W[c + 1] + b[c + 1]));
}

// ---------------------------------------------------------------------------
// Node-side GEMM: Uh, Vh, Bh, Ch = h @ {U,V,B,C} + bias   (A-fragments reused)
// ---------------------------------------------------------------------------
__global__ void k_node_gemm(const bf16_t* __restrict__ h, const bf16_t* __restrict__ Wl,
                            const float* __restrict__ bU, const float* __restrict__ bV,
                            const float* __restrict__ bB, const float* __restrict__ bC,
                            bf16_t* __restrict__ Uh, bf16_t* __restrict__ Vh,
                            bf16_t* __restrict__ Bh, bf16_t* __restrict__ Ch) {
    int wv = threadIdx.x >> 5, lane = threadIdx.x & 31;
    int m0 = (blockIdx.x * 8 + wv) * 16;
    if (m0 >= N_NODES) return;              // whole-wave guard; tiles are full
    const bf16_t* Ws[4]  = { Wl, Wl + 4096, Wl + 3 * 4096, Wl + 4 * 4096 }; // U,V,B,C
    const float*  bs[4]  = { bU, bV, bB, bC };
    bf16_t*       outs[4]= { Uh, Vh, Bh, Ch };
    int rbase = m0 + 8 * (lane >> 4);
    v16bf a0 = load_a_bf(h, DIM, m0, 0);
    v16bf a1 = load_a_bf(h, DIM, m0, 32);
#pragma unroll
    for (int w = 0; w < 4; ++w) {
        v8f acc[4] = {};
#pragma unroll
        for (int nt = 0; nt < 4; ++nt) {
            acc[nt] = __builtin_amdgcn_wmma_f32_16x16x32_bf16(
                false, a0, false, load_b_swz(Ws[w], nt, 0), (short)0, acc[nt], false, false);
            acc[nt] = __builtin_amdgcn_wmma_f32_16x16x32_bf16(
                false, a1, false, load_b_swz(Ws[w], nt, 1), (short)0, acc[nt], false, false);
        }
#pragma unroll
        for (int nt = 0; nt < 4; ++nt) {
            int col = nt * 16 + (lane & 15);
            float bia = bs[w][col];
            bf16_t* o = outs[w] + (long)rbase * DIM + col;   // base + r*128B imm
#pragma unroll
            for (int r = 0; r < 8; ++r) o[r * DIM] = f2bf(acc[nt][r] + bia);
        }
    }
}

// ---------------------------------------------------------------------------
// Edge-side GEMM fused with gathers + BN-stat accumulation:
//   t_e = e @ A + bA + Bh[src] + Ch[dst];  stats += (sum, sumsq) per column
// ---------------------------------------------------------------------------
__global__ void k_edge_gemm(const bf16_t* __restrict__ e, const bf16_t* __restrict__ WA,
                            const float* __restrict__ bA,
                            const bf16_t* __restrict__ Bh, const bf16_t* __restrict__ Ch,
                            const int* __restrict__ src, const int* __restrict__ dst,
                            bf16_t* __restrict__ tE, float* __restrict__ stats) {
    __shared__ float sSum[64], sSsq[64];
    int tid = threadIdx.x;
    if (tid < 64) { sSum[tid] = 0.f; sSsq[tid] = 0.f; }
    __syncthreads();
    int wv = tid >> 5, lane = tid & 31;
    int m0 = (blockIdx.x * 8 + wv) * 16;    // grid tiles exactly: no guards

    // Hoist this lane's 8 gather indices; prefetch while the WMMAs run.
    int rbase = m0 + 8 * (lane >> 4);
    int sIdx[8], dIdx[8];
#pragma unroll
    for (int r = 0; r < 8; ++r) { sIdx[r] = src[rbase + r]; dIdx[r] = dst[rbase + r]; }
    __builtin_prefetch(Bh + (long)sIdx[0] * DIM, 0, 3);   // global_prefetch_b8
    __builtin_prefetch(Ch + (long)dIdx[0] * DIM, 0, 3);

    v8f acc[4] = {};
    gemm16x64(e, DIM, m0, WA, acc);
#pragma unroll
    for (int nt = 0; nt < 4; ++nt) {
        int col = nt * 16 + (lane & 15);
        float bia = bA[col];
        bf16_t* o = tE + (long)rbase * DIM + col;
        float pS = 0.f, pQ = 0.f;            // per-lane register partials
#pragma unroll
        for (int r = 0; r < 8; ++r) {
            float v = acc[nt][r] + bia
                    + bf2f(Bh[(long)sIdx[r] * DIM + col])
                    + bf2f(Ch[(long)dIdx[r] * DIM + col]);
            o[r * DIM] = f2bf(v);
            pS += v; pQ += v * v;
        }
        atomicAdd(&sSum[col], pS);           // 4 ds_add per lane (was 32)
        atomicAdd(&sSsq[col], pQ);
    }
    __syncthreads();
    if (tid < 64) {
        atomicAdd(&stats[tid], sSum[tid]);
        atomicAdd(&stats[64 + tid], sSsq[tid]);
    }
}

// msg = sigmoid(e) * Vh[dst];  agg[src] += msg (f32);  cnt[src] += 1
// Processes bf16 pairs: one b32 load per tensor per 2 elements.
__global__ void k_scatter(const unsigned* __restrict__ e2, const unsigned* __restrict__ Vh2,
                          const int* __restrict__ src, const int* __restrict__ dst,
                          float* __restrict__ agg, float* __restrict__ cnt) {
    size_t p = (size_t)blockIdx.x * blockDim.x + threadIdx.x;   // pair index
    if (p >= (size_t)N_EDGES * DIM / 2) return;
    size_t i = p * 2;
    int row = (int)(i >> 6), c = (int)(i & 63);
    int s = src[row], d = dst[row];
    unsigned ep = e2[p];
    unsigned vp = Vh2[((long)d * DIM + c) >> 1];
    float m0v = sigf(bflo(ep)) * bflo(vp);
    float m1v = sigf(bfhi(ep)) * bfhi(vp);
    atomicAdd(&agg[(long)s * DIM + c], m0v);
    atomicAdd(&agg[(long)s * DIM + c + 1], m1v);
    if (c == 0) atomicAdd(&cnt[s], 1.0f);
}

// t_h = Uh + agg / max(cnt,1)  (in-place over Uh, bf16) + BN stats (f32)
__global__ void k_hstats(bf16_t* __restrict__ Uh, const float* __restrict__ agg,
                         const float* __restrict__ cnt, float* __restrict__ stats) {
    __shared__ float sSum[64], sSsq[64];
    int tid = threadIdx.x;
    if (tid < 64) { sSum[tid] = 0.f; sSsq[tid] = 0.f; }
    __syncthreads();
    size_t idx = (size_t)blockIdx.x * blockDim.x + tid;
    if (idx < (size_t)N_NODES * DIM) {
        int row = (int)(idx >> 6), c = (int)(idx & 63);
        float t = bf2f(Uh[idx]) + agg[idx] / fmaxf(cnt[row], 1.0f);
        Uh[idx] = f2bf(t);
        atomicAdd(&sSum[c], t);
        atomicAdd(&sSsq[c], t * t);
    }
    __syncthreads();
    if (tid < 64) {
        atomicAdd(&stats[tid], sSum[tid]);
        atomicAdd(&stats[64 + tid], sSsq[tid]);
    }
}

// tgt += silu( (t - mu) * rsqrt(var+eps) * g + b )  -- bf16 pairs in-place
__global__ void k_apply(unsigned* __restrict__ tgt2, const unsigned* __restrict__ t2,
                        const float* __restrict__ stats, const float* __restrict__ g,
                        const float* __restrict__ b, float invn, size_t npairs) {
    size_t p = (size_t)blockIdx.x * blockDim.x + threadIdx.x;
    if (p >= npairs) return;
    size_t i = p * 2;
    int c = (int)(i & 63);
    float mu0 = stats[c] * invn,     mu1 = stats[c + 1] * invn;
    float v0  = fmaxf(stats[64 + c] * invn - mu0 * mu0, 0.f);
    float v1  = fmaxf(stats[64 + c + 1] * invn - mu1 * mu1, 0.f);
    unsigned tp = t2[p], gp = tgt2[p];
    float x0 = (bflo(tp) - mu0) * rsqrtf(v0 + EPS_BN) * g[c]     + b[c];
    float x1 = (bfhi(tp) - mu1) * rsqrtf(v1 + EPS_BN) * g[c + 1] + b[c + 1];
    tgt2[p] = pack_bf2(bflo(gp) + siluf(x0), bfhi(gp) + siluf(x1));
}

// ---------------------------------------------------------------------------
// Fused edge MLP head: silu(e@W1+b1) -> silu(.@W2+b2) -> sigmoid(.@w3+b3)
// LDS round-trip (bf16 tile) converts WMMA D-layout back to A-layout.
// ---------------------------------------------------------------------------
__global__ void k_fe(const bf16_t* __restrict__ e,
                     const bf16_t* __restrict__ W1, const float* __restrict__ b1,
                     const bf16_t* __restrict__ W2, const float* __restrict__ b2,
                     const float* __restrict__ W3, const float* __restrict__ b3,
                     float* __restrict__ out) {
    __shared__ __align__(16) bf16_t zt[8][16][72];  // per-wave 16x64 tile (+pad)
    int tid = threadIdx.x, wv = tid >> 5, lane = tid & 31;
    int m0 = (blockIdx.x * 8 + wv) * 16;            // exact tiling: no guards

    v8f acc[4] = {};
    gemm16x64(e, DIM, m0, W1, acc);
#pragma unroll
    for (int nt = 0; nt < 4; ++nt) {
        int col = nt * 16 + (lane & 15);
        float bia = b1[col];
#pragma unroll
        for (int r = 0; r < 8; ++r)
            zt[wv][r + 8 * (lane >> 4)][col] = f2bf(siluf(acc[nt][r] + bia));
    }
    __syncthreads();

    v8f acc2[4] = {};
    gemm16x64(&zt[wv][0][0], 72, 0, W2, acc2);
    __syncthreads();                          // all zt reads done before overwrite
#pragma unroll
    for (int nt = 0; nt < 4; ++nt) {
        int col = nt * 16 + (lane & 15);
        float bia = b2[col];
#pragma unroll
        for (int r = 0; r < 8; ++r)
            zt[wv][r + 8 * (lane >> 4)][col] = f2bf(siluf(acc2[nt][r] + bia));
    }
    __syncthreads();

    // Final 64-dot per row: split across lane halves, combine via shfl_xor(16).
    int half = lane >> 4, rrow = lane & 15;
    float s = 0.f;
#pragma unroll
    for (int c = 0; c < 32; ++c)
        s += bf2f(zt[wv][rrow][half * 32 + c]) * W3[half * 32 + c];
    s += __shfl_xor(s, 16, 32);
    if (lane < 16) out[m0 + rrow] = sigf(s + b3[0]);
}

// ---------------------------------------------------------------------------
// Host launcher
// ---------------------------------------------------------------------------
extern "C" void kernel_launch(void* const* d_in, const int* in_sizes, int n_in,
                              void* d_out, int out_size, void* d_ws, size_t ws_size,
                              hipStream_t stream) {
    (void)in_sizes; (void)n_in; (void)out_size; (void)ws_size;

    // --- inputs (setup_inputs dict insertion order, leaves flattened) ------
    const float* x    = (const float*)d_in[0];
    const float* ea   = (const float*)d_in[1];
    const int*   eidx = (const int*)d_in[2];
    const int* src = eidx;
    const int* dst = eidx + N_EDGES;
    const float* hW = (const float*)d_in[3];
    const float* hb = (const float*)d_in[4];
    const float* eW = (const float*)d_in[5];
    const float* eb = (const float*)d_in[6];
    // layers: 7 + l*14 : U(W,b) V(W,b) A(W,b) B(W,b) C(W,b) hg hb eg eb
    // fe: base 49: W1 b1 W2 b2 W3 b3
    const float* feW1 = (const float*)d_in[49];
    const float* feB1 = (const float*)d_in[50];
    const float* feW2 = (const float*)d_in[51];
    const float* feB2 = (const float*)d_in[52];
    const float* feW3 = (const float*)d_in[53];
    const float* feB3 = (const float*)d_in[54];

    // --- workspace layout --------------------------------------------------
    const size_t nh = (size_t)N_NODES * DIM;   // 3.2M
    const size_t ne = (size_t)N_EDGES * DIM;   // 76.8M
    char* w = (char*)d_ws;
    bf16_t* wbf = (bf16_t*)w;      w += 17 * 4096 * sizeof(bf16_t);
    float* agg  = (float*)w;       w += nh * sizeof(float);   // zeroed block start
    float* cnt  = (float*)w;       w += (size_t)N_NODES * sizeof(float);
    float* stH  = (float*)w;       w += 128 * sizeof(float);
    float* stE  = (float*)w;       w += 128 * sizeof(float);
    bf16_t* h   = (bf16_t*)w;      w += nh * sizeof(bf16_t);
    bf16_t* Uh  = (bf16_t*)w;      w += nh * sizeof(bf16_t);  // becomes t_h
    bf16_t* Vh  = (bf16_t*)w;      w += nh * sizeof(bf16_t);
    bf16_t* Bh  = (bf16_t*)w;      w += nh * sizeof(bf16_t);
    bf16_t* Ch  = (bf16_t*)w;      w += nh * sizeof(bf16_t);
    bf16_t* e   = (bf16_t*)w;      w += ne * sizeof(bf16_t);
    bf16_t* tE  = (bf16_t*)w;      w += ne * sizeof(bf16_t);
    const size_t zero_bytes = (nh + (size_t)N_NODES + 256) * sizeof(float);

    // --- weight conversion f32 -> swizzled bf16 B-fragments ----------------
    for (int l = 0; l < 3; ++l)
        for (int i = 0; i < 5; ++i)   // U,V,A,B,C
            k_cvt_swz<<<1, 256, 0, stream>>>(
                (const float*)d_in[7 + l * 14 + i * 2], wbf + (l * 5 + i) * 4096);
    k_cvt_swz<<<1, 256, 0, stream>>>(feW1, wbf + 15 * 4096);
    k_cvt_swz<<<1, 256, 0, stream>>>(feW2, wbf + 16 * 4096);

    // --- input projections -------------------------------------------------
    k_init_h<<<(unsigned)((nh + 255) / 256), 256, 0, stream>>>(x, hW, hb, h);
    k_init_e<<<(unsigned)((ne / 2 + 255) / 256), 256, 0, stream>>>(ea, eW, eb, (unsigned*)e);

    const int nodeBlocks = (int)(((N_NODES + 15) / 16 + 7) / 8);   // 391
    const int edgeBlocks = (int)(N_EDGES / 128);                   // 9375 exact
    const unsigned gridNH  = (unsigned)((nh + 255) / 256);
    const unsigned gridNHp = (unsigned)((nh / 2 + 255) / 256);
    const unsigned gridNEp = (unsigned)((ne / 2 + 255) / 256);

    for (int l = 0; l < 3; ++l) {
        const float* Ub  = (const float*)d_in[7 + l * 14 + 1];
        const float* Vb  = (const float*)d_in[7 + l * 14 + 3];
        const float* Ab  = (const float*)d_in[7 + l * 14 + 5];
        const float* Bb  = (const float*)d_in[7 + l * 14 + 7];
        const float* Cb  = (const float*)d_in[7 + l * 14 + 9];
        const float* hg  = (const float*)d_in[7 + l * 14 + 10];
        const float* hbb = (const float*)d_in[7 + l * 14 + 11];
        const float* eg  = (const float*)d_in[7 + l * 14 + 12];
        const float* ebb = (const float*)d_in[7 + l * 14 + 13];
        const bf16_t* Wl = wbf + l * 5 * 4096;

        hipMemsetAsync(agg, 0, zero_bytes, stream);  // agg, cnt, stH, stE
        k_node_gemm<<<nodeBlocks, 256, 0, stream>>>(h, Wl, Ub, Vb, Bb, Cb,
                                                    Uh, Vh, Bh, Ch);
        k_edge_gemm<<<edgeBlocks, 256, 0, stream>>>(e, Wl + 2 * 4096, Ab,
                                                    Bh, Ch, src, dst, tE, stE);
        k_scatter<<<gridNEp, 256, 0, stream>>>((const unsigned*)e, (const unsigned*)Vh,
                                               src, dst, agg, cnt);
        k_hstats<<<gridNH, 256, 0, stream>>>(Uh, agg, cnt, stH);
        k_apply<<<gridNHp, 256, 0, stream>>>((unsigned*)h, (const unsigned*)Uh, stH,
                                             hg, hbb, 1.0f / (float)N_NODES, nh / 2);
        k_apply<<<gridNEp, 256, 0, stream>>>((unsigned*)e, (const unsigned*)tE, stE,
                                             eg, ebb, 1.0f / (float)N_EDGES, ne / 2);
    }

    k_fe<<<edgeBlocks, 256, 0, stream>>>(e, wbf + 15 * 4096, feB1,
                                         wbf + 16 * 4096, feB2,
                                         feW3, feB3, (float*)d_out);
}
